// Denoiser_21406117003980
// MI455X (gfx1250) — matrix operand order repaired and checked by reading, not compiled, not run on hardware
//
#include <hip/hip_runtime.h>
#include <hip/hip_bf16.h>
#include <cstdint>

// ---------------------------------------------------------------------------
// MI455X (gfx1250) SSM U-Net denoiser.
// Frequency-domain SSM long-conv via Stockham FFT + WMMA f32 16x16x4 GEMMs
// (B panel staged in LDS via async global->LDS copies, ASYNCcnt-tracked),
// bottleneck cross-attention projections via WMMA f32_16x16x32_f16.
// wave32 / WGP aware: 256-thread blocks = 8 waves, 1 WMMA 16x16 tile per wave.
// ---------------------------------------------------------------------------

typedef __attribute__((ext_vector_type(2)))  float    v2f;
typedef __attribute__((ext_vector_type(8)))  float    v8f;
typedef __attribute__((ext_vector_type(16))) _Float16 v16h;

// -------------------- CDNA5 async global->LDS helpers ----------------------
__device__ __forceinline__ void async_copy_b32(float* lds_dst, const float* gsrc)
{
#if defined(__gfx1250__)
    unsigned ldsoff = (unsigned)(uintptr_t)lds_dst;   // low 32 bits = LDS offset
    asm volatile("global_load_async_to_lds_b32 %0, %1, off"
                 :: "v"(ldsoff), "v"(gsrc) : "memory");
#else
    *lds_dst = *gsrc;
#endif
}

__device__ __forceinline__ void async_join()
{
#if defined(__gfx1250__)
    asm volatile("s_wait_asynccnt 0" ::: "memory");
#endif
}

// ------------------------------- FFT ---------------------------------------
// One Stockham radix-2 stage over planar complex rows (ping-pong buffers).
__global__ void fft_stage(const float* __restrict__ xr, const float* __restrict__ xi,
                          float* __restrict__ yr, float* __restrict__ yi,
                          int n, int mshift, float sgn, float scale)
{
    int half = n >> 1;
    int tid = blockIdx.x * blockDim.x + threadIdx.x;
    if (tid >= half) return;
    size_t roff = (size_t)blockIdx.y * (size_t)n;
    int m = 1 << mshift;
    int l = half >> mshift;
    int q = tid & (m - 1);
    int p = tid >> mshift;
    size_t ia = roff + (size_t)q + ((size_t)p << mshift);
    size_t ib = ia + (size_t)half;
    float arv = xr[ia], aiv = xi[ia];
    float brv = xr[ib], biv = xi[ib];
    float ang = sgn * 3.14159265358979323846f * (float)p / (float)l;
    float sw, cw;
    __sincosf(ang, &sw, &cw);
    float sr = arv + brv, si = aiv + biv;
    float dr = arv - brv, di = aiv - biv;
    size_t o0 = roff + (size_t)q + ((size_t)p << (mshift + 1));
    size_t o1 = o0 + (size_t)m;
    yr[o0] = sr * scale;
    yi[o0] = si * scale;
    yr[o1] = (dr * cw - di * sw) * scale;
    yi[o1] = (dr * sw + di * cw) * scale;
}

__global__ void load_complex(const float* __restrict__ x, float* __restrict__ re,
                             float* __restrict__ im, int L, int n2)
{
    int col = blockIdx.x * blockDim.x + threadIdx.x;
    if (col >= n2) return;
    size_t row = blockIdx.y;
    re[row * (size_t)n2 + col] = (col < L) ? x[row * (size_t)L + col] : 0.0f;
    im[row * (size_t)n2 + col] = 0.0f;
}

__global__ void take_real(const float* __restrict__ re, float* __restrict__ out, int n2, int L)
{
    int col = blockIdx.x * blockDim.x + threadIdx.x;
    if (col >= L) return;
    size_t row = blockIdx.y;
    out[row * (size_t)L + col] = re[row * (size_t)n2 + col];
}

__global__ void fill0(float* __restrict__ p, long long n)
{
    long long i = (long long)blockIdx.x * blockDim.x + threadIdx.x;
    if (i < n) p[i] = 0.0f;
}

__global__ void addv(const float* __restrict__ a, const float* __restrict__ b,
                     float* __restrict__ y, long long n)
{
    long long i = (long long)blockIdx.x * blockDim.x + threadIdx.x;
    if (i < n) y[i] = a[i] + b[i];
}

// --------------------------- SSM parameter prep ----------------------------
// K[n,l] = exp(-dt*softplus(A0)*l) * cos(dt*A1*l)
__global__ void build_Kt(const float* __restrict__ A2, const float* __restrict__ ldt,
                         float* __restrict__ Kt, int L)
{
    int l = blockIdx.x * blockDim.x + threadIdx.x;
    if (l >= L) return;
    int n = blockIdx.y;
    float dt = __expf(ldt[n]);
    float sp = log1pf(__expf(A2[2 * n]));
    float ar = -dt * sp;
    float ai = dt * A2[2 * n + 1];
    float t = (float)l;
    Kt[(size_t)n * L + l] = __expf(ar * t) * __cosf(ai * t);
}

__global__ void build_Kpad(const float* __restrict__ A2, const float* __restrict__ ldt,
                           float* __restrict__ kr, float* __restrict__ ki, int L, int n2)
{
    int f = blockIdx.x * blockDim.x + threadIdx.x;
    if (f >= n2) return;
    int n = blockIdx.y;
    float v = 0.0f;
    if (f < L) {
        float dt = __expf(ldt[n]);
        float sp = log1pf(__expf(A2[2 * n]));
        float t = (float)f;
        v = __expf(-dt * sp * t) * __cosf(dt * A2[2 * n + 1] * t);
    }
    kr[(size_t)n * n2 + f] = v;
    ki[(size_t)n * n2 + f] = 0.0f;
}

// W[(di*c+ci)*256 + n] = C[di,n] * B[n,ci] * dt[n]
__global__ void build_W(const float* __restrict__ C, const float* __restrict__ Bp,
                        const float* __restrict__ ldt, float* __restrict__ W, int c, int d)
{
    int i = blockIdx.x * blockDim.x + threadIdx.x;
    int tot = d * c * 256;
    if (i >= tot) return;
    int n = i & 255;
    int dc = i >> 8;
    int di = dc / c, ci = dc % c;
    W[i] = C[di * 256 + n] * Bp[n * c + ci] * __expf(ldt[n]);
}

__global__ void build_Bh(const float* __restrict__ Bp, const float* __restrict__ ldt,
                         float* __restrict__ Bh, int c)
{
    int i = blockIdx.x * blockDim.x + threadIdx.x;
    if (i >= 256 * c) return;
    int n = i / c;
    Bh[i] = Bp[i] * __expf(ldt[n]);
}

// out_f[b,di,f] = sum_ci Mf[di,ci,f] * Xf[b,ci,f]   (complex)
__global__ void small_apply(const float* __restrict__ xr, const float* __restrict__ xi,
                            const float* __restrict__ mr, const float* __restrict__ mi,
                            float* __restrict__ orr, float* __restrict__ oii,
                            int c, int d, int n2)
{
    int f = blockIdx.x * blockDim.x + threadIdx.x;
    if (f >= n2) return;
    int di = blockIdx.y, b = blockIdx.z;
    float ar = 0.0f, ai = 0.0f;
    for (int ci = 0; ci < c; ++ci) {
        size_t xo = ((size_t)(b * c + ci)) * n2 + f;
        size_t mo = ((size_t)(di * c + ci)) * n2 + f;
        float xrv = xr[xo], xiv = xi[xo];
        float mrv = mr[mo], miv = mi[mo];
        ar += xrv * mrv - xiv * miv;
        ai += xrv * miv + xiv * mrv;
    }
    size_t oo = ((size_t)(b * d + di)) * n2 + f;
    orr[oo] = ar;
    oii[oo] = ai;
}

// Y[b,n,f] *= Kf[n,f]  (complex, in place)
__global__ void cmul_Kf(float* __restrict__ yr, float* __restrict__ yi,
                        const float* __restrict__ kr, const float* __restrict__ ki, int n2)
{
    int f = blockIdx.x * blockDim.x + threadIdx.x;
    if (f >= n2) return;
    int n = blockIdx.y, b = blockIdx.z;
    size_t yo = ((size_t)(b * 256 + n)) * n2 + f;
    size_t ko = (size_t)n * n2 + f;
    float a = yr[yo], bb = yi[yo];
    float cr = kr[ko], ci = ki[ko];
    yr[yo] = a * cr - bb * ci;
    yi[yo] = a * ci + bb * cr;
}

// -------------------------- WMMA f32 strided GEMM --------------------------
// D[m,n] (+)= alpha * sum_k A[m,k]*B[k,n] + bias[n]; fully strided, batched z.
// One wave = one 16x16 tile; 8 waves per 256-thread block (wave32).
// Requirements (all call sites satisfy): K <= GEMM_MAXK. M%16 / N ragged OK.
// B panel (K x 16, shared by all waves of the block) is staged into LDS via
// async global->LDS copies; inner loop is branch-free:
//   2x global_load_b32 (A) + 1x ds_load_b64 (B) + v_wmma_f32_16x16x4_f32.
#define GEMM_MAXK 256

__global__ void wmma_gemm_f32(const float* __restrict__ A, long sAm, long sAk, long sAz,
                              const float* __restrict__ Bm, long sBk, long sBn, long sBz,
                              float* __restrict__ D, long sDm, long sDn, long sDz,
                              const float* __restrict__ bias,
                              int M, int N, int K, float alpha, int accum)
{
    __shared__ float blds[GEMM_MAXK * 16];   // paired layout: (k,c) -> (k>>1)*32 + c*2 + (k&1)
    const int tid = threadIdx.x;
    const int wave = tid >> 5, lane = tid & 31;
    const int n0 = blockIdx.y * 16;
    const int z = blockIdx.z;
    A += (long)z * sAz; Bm += (long)z * sBz; D += (long)z * sDz;

    // cooperative async stage of the B panel (zero-padded to Kp, ragged N)
    const int Kp = (K + 3) & ~3;
    for (int e = tid; e < Kp * 16; e += 256) {
        int k = e >> 4, cl = e & 15;
        int li = ((k >> 1) << 5) + (cl << 1) + (k & 1);
        int col = n0 + cl;
        if (k < K && col < N)
            async_copy_b32(&blds[li], &Bm[(long)k * sBk + (long)col * sBn]);
        else
            blds[li] = 0.0f;
    }
    async_join();
    __syncthreads();

    const int m0 = (blockIdx.x * 8 + wave) * 16;
    if (m0 >= M) return;
    const int hl = lane >> 4;          // lanes 16-31 hold K+2 / M+8 halves
    const int l  = lane & 15;
    int mrow = m0 + l;
    if (mrow > M - 1) mrow = M - 1;    // clamp: rows >= M are never stored
    const float* ap = A + (long)mrow * sAm + (long)(2 * hl) * sAk;
    const long aStep = 4 * sAk;
    const v2f* bl = ((const v2f*)blds) + hl * 16 + l;

    v8f acc = {0.f, 0.f, 0.f, 0.f, 0.f, 0.f, 0.f, 0.f};
    const int Km = K & ~3;
    for (int k0 = 0; k0 < Km; k0 += 4) {
        __builtin_prefetch(ap + 4 * aStep, 0, 1);
        v2f a, b;
        a.x = ap[0];
        a.y = ap[sAk];
        b = *bl;                        // ds_load_b64
#if defined(__gfx1250__) && __has_builtin(__builtin_amdgcn_wmma_f32_16x16x4_f32)
        acc = __builtin_amdgcn_wmma_f32_16x16x4_f32(false, a, false, b,
                                                    (short)0, acc, false, false);
#else
        acc[0] += a.x * b.x + a.y * b.y;   // host-pass placeholder (never executed)
#endif
        ap += aStep;
        bl += 32;
    }
    if (Km < K) {                      // ragged-K tail: B is zero-padded in LDS
        int ka = Km + 2 * hl;
        int k0c = (ka < K) ? ka : K - 1;
        int k1c = (ka + 1 < K) ? ka + 1 : K - 1;
        v2f a, b;
        a.x = A[(long)mrow * sAm + (long)k0c * sAk];
        a.y = A[(long)mrow * sAm + (long)k1c * sAk];
        b = *bl;
#if defined(__gfx1250__) && __has_builtin(__builtin_amdgcn_wmma_f32_16x16x4_f32)
        acc = __builtin_amdgcn_wmma_f32_16x16x4_f32(false, a, false, b,
                                                    (short)0, acc, false, false);
#else
        acc[0] += a.x * b.x + a.y * b.y;
#endif
    }

    const int col = n0 + l;
    if (col < N) {
        float bv = bias ? bias[col] : 0.f;
        for (int j = 0; j < 8; ++j) {
            int row = m0 + j + hl * 8;  // D vgpr j: M=j (lanes 0-15) / M=j+8 (16-31)
            if (row < M) {
                float v = alpha * acc[j] + bv;
                long o = (long)row * sDm + (long)col * sDn;
                D[o] = accum ? D[o] + v : v;
            }
        }
    }
}

// ----------------------- WMMA f16 GEMM (projections) -----------------------
// D[M,N] = A[M,K] @ W[N,K]^T + bias  (row-major f32 in/out, f16 MAC, f32 acc)
__global__ void wmma_gemm_f16w(const float* __restrict__ A, const float* __restrict__ W,
                               const float* __restrict__ bias, float* __restrict__ D,
                               int M, int N, int K)
{
    int wave = threadIdx.x >> 5;
    int lane = threadIdx.x & 31;
    int tm = blockIdx.x * (blockDim.x >> 5) + wave;
    int m0 = tm * 16, n0 = blockIdx.y * 16;
    if (m0 >= M) return;
    int hl = lane >> 4;
    int l  = lane & 15;
    int mrow = m0 + l;
    int col  = n0 + l;
    bool mv = mrow < M;
    bool nv = col < N;
    v8f acc = {0.f, 0.f, 0.f, 0.f, 0.f, 0.f, 0.f, 0.f};
    for (int k0 = 0; k0 < K; k0 += 32) {
        v16h a, b;
        for (int j = 0; j < 16; ++j) {
            int v = j >> 1;
            // 16-bit A 16x32 layout: v0..3 => K 0..7 (+8 for hi half), v4..7 => +16
            int kkA = ((v & 4) ? 16 : 0) + ((v & 3) << 1) + hl * 8 + (j & 1);
            int ka = k0 + kkA;
            a[j] = (_Float16)((mv && ka < K) ? A[(size_t)mrow * K + ka] : 0.f);
            // 16-bit B 32x16 layout: same K order as A: v0..3 => K 0..7 (+8 for hi half), v4..7 => +16
            int kkB = ((v & 4) ? 16 : 0) + ((v & 3) << 1) + hl * 8 + (j & 1);
            int kb = k0 + kkB;
            b[j] = (_Float16)((nv && kb < K) ? W[(size_t)col * K + kb] : 0.f);
        }
#if defined(__gfx1250__)
        acc = __builtin_amdgcn_wmma_f32_16x16x32_f16(false, a, false, b,
                                                     (short)0, acc, false, false);
#else
        acc[0] += (float)a[0] * (float)b[0];   // host-pass placeholder
#endif
    }
    if (nv) {
        float bv = bias ? bias[col] : 0.f;
        for (int j = 0; j < 8; ++j) {
            int row = m0 + j + hl * 8;
            if (row < M) D[(size_t)row * N + col] = acc[j] + bv;
        }
    }
}

// ------------------------------ pointwise ops ------------------------------
__global__ void dwconv3(const float* __restrict__ x, const float* __restrict__ w,
                        const float* __restrict__ bias, float* __restrict__ y, int C, int L)
{
    int t = blockIdx.x * blockDim.x + threadIdx.x;
    if (t >= L) return;
    int c = blockIdx.y, b = blockIdx.z;
    const float* xr = x + ((size_t)b * C + c) * L;
    float w0 = w[c * 3], w1 = w[c * 3 + 1], w2 = w[c * 3 + 2];
    float lft = (t > 0) ? xr[t - 1] : 0.f;
    float mid = xr[t];
    float rgt = (t < L - 1) ? xr[t + 1] : 0.f;
    y[((size_t)b * C + c) * L + t] = w0 * lft + w1 * mid + w2 * rgt + bias[c];
}

// LayerNorm over channels + SiLU (g==null => SiLU only)
__global__ void ln_silu(const float* __restrict__ x, const float* __restrict__ g,
                        const float* __restrict__ b2, float* __restrict__ y, int C, int L)
{
    int t = blockIdx.x * blockDim.x + threadIdx.x;
    if (t >= L) return;
    int bb = blockIdx.y;
    const float* xp = x + (size_t)bb * C * L + t;
    float* yp = y + (size_t)bb * C * L + t;
    if (g) {
        float s = 0.f;
        for (int c = 0; c < C; ++c) s += xp[(size_t)c * L];
        float mu = s / C;
        float vs = 0.f;
        for (int c = 0; c < C; ++c) { float d = xp[(size_t)c * L] - mu; vs += d * d; }
        float rs = rsqrtf(vs / C + 1e-5f);
        for (int c = 0; c < C; ++c) {
            float v = (xp[(size_t)c * L] - mu) * rs * g[c] + b2[c];
            yp[(size_t)c * L] = v / (1.f + __expf(-v));
        }
    } else {
        for (int c = 0; c < C; ++c) {
            float v = xp[(size_t)c * L];
            yp[(size_t)c * L] = v / (1.f + __expf(-v));
        }
    }
}

// out[b,d,t] = sum_{c,rr} x[b,c,t*r+rr] * w[(c*D+d)*r+rr]
__global__ void mix_down(const float* __restrict__ x, const float* __restrict__ w,
                         float* __restrict__ y, int C, int Dd, int Lout, int r)
{
    int t = blockIdx.x * blockDim.x + threadIdx.x;
    if (t >= Lout) return;
    int d = blockIdx.y, b = blockIdx.z;
    int Lin = Lout * r;
    float acc = 0.f;
    for (int c = 0; c < C; ++c) {
        const float* xr = x + ((size_t)b * C + c) * Lin + (size_t)t * r;
        const float* wr = w + ((size_t)c * Dd + d) * r;
        for (int rr = 0; rr < r; ++rr) acc += xr[rr] * wr[rr];
    }
    y[((size_t)b * Dd + d) * Lout + t] = acc;
}

// y[b,d,t*r+rr] = sum_c x[b,c,t]*w[(c*D+d)*r+rr] + skip
__global__ void mix_up_addskip(const float* __restrict__ x, const float* __restrict__ w,
                               const float* __restrict__ skip, float* __restrict__ y,
                               int C, int Dd, int Lin, int r)
{
    int Lout = Lin * r;
    int to = blockIdx.x * blockDim.x + threadIdx.x;
    if (to >= Lout) return;
    int d = blockIdx.y, b = blockIdx.z;
    int t = to / r, rr = to % r;
    float acc = 0.f;
    for (int c = 0; c < C; ++c)
        acc += x[((size_t)b * C + c) * Lin + t] * w[((size_t)c * Dd + d) * r + rr];
    size_t oo = ((size_t)b * Dd + d) * Lout + to;
    y[oo] = acc + (skip ? skip[oo] : 0.f);
}

__global__ void interp1d_k(const float* __restrict__ x, float* __restrict__ y, int S, int T)
{
    int t = blockIdx.x * blockDim.x + threadIdx.x;
    if (t >= T) return;
    size_t row = blockIdx.y;
    float coord = ((float)t + 0.5f) * ((float)S / (float)T) - 0.5f;
    coord = fminf(fmaxf(coord, 0.f), (float)(S - 1));
    int lo = (int)floorf(coord);
    int hi = min(lo + 1, S - 1);
    float w = coord - (float)lo;
    const float* xr = x + row * (size_t)S;
    y[row * (size_t)T + t] = xr[lo] * (1.f - w) + xr[hi] * w;
}

// bottleneck reshuffle: x (B,256,k) -> seq (B, 256*k) per flatten of (k,B,256)
__global__ void pack_seq(const float* __restrict__ x, float* __restrict__ seq, int Bb, int k)
{
    long long i = (long long)blockIdx.x * blockDim.x + threadIdx.x;
    long long tot = (long long)Bb * 256 * k;
    if (i >= tot) return;
    long long kk = i / ((long long)Bb * 256);
    long long rem = i % ((long long)Bb * 256);
    int bb = (int)(rem / 256), ch = (int)(rem % 256);
    seq[i] = x[((size_t)bb * 256 + ch) * k + kk];
}

__global__ void unpack_seq(const float* __restrict__ seq, float* __restrict__ x, int Bb, int k)
{
    long long i = (long long)blockIdx.x * blockDim.x + threadIdx.x;
    long long tot = (long long)Bb * 256 * k;
    if (i >= tot) return;
    long long kk = i / ((long long)Bb * 256);
    long long rem = i % ((long long)Bb * 256);
    int bb = (int)(rem / 256), ch = (int)(rem % 256);
    x[((size_t)bb * 256 + ch) * k + kk] = seq[i];
}

// 4-head cross attention core: one wave (32 lanes) per (b, h, query row)
__global__ void mha_attn(const float* __restrict__ q, const float* __restrict__ k,
                         const float* __restrict__ v, float* __restrict__ o,
                         int T, int E, int H, int Dh)
{
    int b = blockIdx.z, h = blockIdx.y, qi = blockIdx.x;
    int lane = threadIdx.x;
    __shared__ float sc[128];
    const float* qp = q + ((size_t)(b * T + qi)) * E + h * Dh;
    float scale = rsqrtf((float)Dh);
    for (int j = lane; j < T; j += 32) {
        const float* kp = k + ((size_t)(b * T + j)) * E + h * Dh;
        float s = 0.f;
        for (int e = 0; e < Dh; ++e) s += qp[e] * kp[e];
        sc[j] = s * scale;
    }
    __syncthreads();
    float mx = -1e30f;
    for (int j = lane; j < T; j += 32) mx = fmaxf(mx, sc[j]);
    for (int off = 16; off; off >>= 1) mx = fmaxf(mx, __shfl_xor(mx, off, 32));
    float sm = 0.f;
    for (int j = lane; j < T; j += 32) { float e = __expf(sc[j] - mx); sc[j] = e; sm += e; }
    for (int off = 16; off; off >>= 1) sm += __shfl_xor(sm, off, 32);
    __syncthreads();
    float inv = 1.f / sm;
    if (lane < Dh) {
        float acc = 0.f;
        for (int j = 0; j < T; ++j)
            acc += sc[j] * v[((size_t)(b * T + j)) * E + h * Dh + lane];
        o[((size_t)(b * T + qi)) * E + h * Dh + lane] = acc * inv;
    }
}

// ------------------------------ host drivers -------------------------------
struct Planes { float* re; float* im; };

static Planes fft_run(hipStream_t st, float* ar, float* ai, float* br, float* bi,
                      int rows, int n, bool inv)
{
    float sgn = inv ? 1.0f : -1.0f;
    int half = n >> 1;
    dim3 blk(256), grd((half + 255) / 256, rows);
    int mshift = 0;
    for (int m = 1; m < n; m <<= 1, ++mshift) {
        float scale = (inv && (m << 1) == n) ? (1.0f / (float)n) : 1.0f;
        fft_stage<<<grd, blk, 0, st>>>(ar, ai, br, bi, n, mshift, sgn, scale);
        float* t;
        t = ar; ar = br; br = t;
        t = ai; ai = bi; bi = t;
    }
    Planes p; p.re = ar; p.im = ai; return p;
}

static void gemm_f32(hipStream_t st,
                     const float* A, long sAm, long sAk, long sAz,
                     const float* B, long sBk, long sBn, long sBz,
                     float* D, long sDm, long sDn, long sDz,
                     const float* bias, int M, int N, int K, int Z,
                     float alpha, int accum)
{
    dim3 grd(((M + 15) / 16 + 7) / 8, (N + 15) / 16, Z);
    wmma_gemm_f32<<<grd, dim3(256), 0, st>>>(A, sAm, sAk, sAz, B, sBk, sBn, sBz,
                                             D, sDm, sDn, sDz, bias, M, N, K, alpha, accum);
}

static void run_ssm(hipStream_t st, char* scr,
                    const float* x, float* out, int Bb, int c, int L,
                    const float* Ap, const float* Bp, const float* Cp, const float* ldt)
{
    int d = c;
    int n2 = 2 * L;
    size_t off = 0;
    auto alloc = [&](size_t nf) -> float* {
        float* p = (float*)(scr + off);
        off += ((nf * sizeof(float) + 255) & ~(size_t)255);
        return p;
    };
    int rcx = Bb * c;
    float* xr = alloc((size_t)rcx * n2); float* xi = alloc((size_t)rcx * n2);
    float* txr = alloc((size_t)rcx * n2); float* txi = alloc((size_t)rcx * n2);
    load_complex<<<dim3((n2 + 255) / 256, rcx), 256, 0, st>>>(x, xr, xi, L, n2);
    Planes Xf = fft_run(st, xr, xi, txr, txi, rcx, n2, false);

    int rdo = Bb * d;
    float* orr = alloc((size_t)rdo * n2); float* oii = alloc((size_t)rdo * n2);
    float* tor_ = alloc((size_t)rdo * n2); float* toi = alloc((size_t)rdo * n2);

    if ((long)c * d <= 256) {
        // small-channel path: time-domain kernel, then spectrum
        float* Kt = alloc((size_t)256 * L);
        build_Kt<<<dim3((L + 255) / 256, 256), 256, 0, st>>>(Ap, ldt, Kt, L);
        int DC = d * c;
        float* W = alloc((size_t)DC * 256);
        build_W<<<(DC * 256 + 255) / 256, 256, 0, st>>>(Cp, Bp, ldt, W, c, d);
        float* kr = alloc((size_t)DC * n2); float* ki = alloc((size_t)DC * n2);
        float* tkr = alloc((size_t)DC * n2); float* tki = alloc((size_t)DC * n2);
        long long knum = (long long)DC * n2;
        fill0<<<(int)((knum + 255) / 256), 256, 0, st>>>(kr, knum);
        fill0<<<(int)((knum + 255) / 256), 256, 0, st>>>(ki, knum);
        // ker[dc,l] = sum_n Kt[n,l] * W[dc,n]  (WMMA GEMM, M=L, N=DC, K=256)
        gemm_f32(st, Kt, 1, (long)L, 0, W, 1, 256, 0, kr, 1, (long)n2, 0,
                 nullptr, L, DC, 256, 1, 1.0f, 0);
        Planes Kf = fft_run(st, kr, ki, tkr, tki, DC, n2, false);
        small_apply<<<dim3((n2 + 255) / 256, d, Bb), 256, 0, st>>>(
            Xf.re, Xf.im, Kf.re, Kf.im, orr, oii, c, d, n2);
    } else {
        // frequency path: project to 256 states, scale by K_f, project back
        float* kr = alloc((size_t)256 * n2); float* ki = alloc((size_t)256 * n2);
        float* tkr = alloc((size_t)256 * n2); float* tki = alloc((size_t)256 * n2);
        build_Kpad<<<dim3((n2 + 255) / 256, 256), 256, 0, st>>>(Ap, ldt, kr, ki, L, n2);
        Planes Kf = fft_run(st, kr, ki, tkr, tki, 256, n2, false);
        float* Bh = alloc((size_t)256 * c);
        build_Bh<<<(256 * c + 255) / 256, 256, 0, st>>>(Bp, ldt, Bh, c);
        float* yrp = alloc((size_t)Bb * 256 * n2);
        float* yip = alloc((size_t)Bb * 256 * n2);
        // Y[b,n,f] = sum_c Xf[b,c,f]*Bh[n,c]   M=n2, N=256, K=c, Z=B
        gemm_f32(st, Xf.re, 1, (long)n2, (long)c * n2, Bh, 1, (long)c, 0,
                 yrp, 1, (long)n2, (long)256 * n2, nullptr, n2, 256, c, Bb, 1.0f, 0);
        gemm_f32(st, Xf.im, 1, (long)n2, (long)c * n2, Bh, 1, (long)c, 0,
                 yip, 1, (long)n2, (long)256 * n2, nullptr, n2, 256, c, Bb, 1.0f, 0);
        cmul_Kf<<<dim3((n2 + 255) / 256, 256, Bb), 256, 0, st>>>(yrp, yip, Kf.re, Kf.im, n2);
        // out[b,d,f] = sum_n Y[b,n,f]*C[d,n]  M=n2, N=d, K=256, Z=B
        gemm_f32(st, yrp, 1, (long)n2, (long)256 * n2, Cp, 1, 256, 0,
                 orr, 1, (long)n2, (long)d * n2, nullptr, n2, d, 256, Bb, 1.0f, 0);
        gemm_f32(st, yip, 1, (long)n2, (long)256 * n2, Cp, 1, 256, 0,
                 oii, 1, (long)n2, (long)d * n2, nullptr, n2, d, 256, Bb, 1.0f, 0);
    }
    Planes Of = fft_run(st, orr, oii, tor_, toi, rdo, n2, true);
    take_real<<<dim3((L + 255) / 256, rdo), 256, 0, st>>>(Of.re, out, n2, L);
}

static void run_block(hipStream_t st, char* scr,
                      const float* xin, float* xout, float* convt,
                      int Bb, int c, int L, bool act,
                      const float* cw, const float* cb,
                      const float* lg, const float* lb,
                      const float* Ap, const float* Bp, const float* Cp, const float* ldt)
{
    const float* cur = xin;
    if (cw) {
        dwconv3<<<dim3((L + 255) / 256, c, Bb), 256, 0, st>>>(xin, cw, cb, convt, c, L);
        cur = convt;
    }
    run_ssm(st, scr, cur, xout, Bb, c, L, Ap, Bp, Cp, ldt);
    if (act)
        ln_silu<<<dim3((L + 255) / 256, Bb), 256, 0, st>>>(xout, lg, lb, xout, c, L);
}

// ------------------------------ entry point --------------------------------
extern "C" void kernel_launch(void* const* d_in, const int* in_sizes, int n_in,
                              void* d_out, int out_size, void* d_ws, size_t ws_size,
                              hipStream_t stream)
{
    (void)in_sizes; (void)out_size; (void)ws_size;
    const int Bb = 4;
    const int CH[7] = {1, 16, 32, 64, 96, 128, 256};
    const int RS[6] = {4, 4, 2, 2, 2, 2};
    int Ls[7];
    Ls[0] = 65536;
    for (int i = 0; i < 6; ++i) Ls[i + 1] = Ls[i] / RS[i];

    auto F = [&](int i) -> const float* {
        return (i >= 0 && i < n_in) ? (const float*)d_in[i] : nullptr;
    };
    const float* x_in = F(0);
    const float* vf   = F(1);

    // params flattened after input/vf: dicts sorted by key, lists in order.
    int ix = 2;
    const float *dA[6], *dB[6], *dC[6], *dcb[6], *dcw[6], *dlb[6], *dlg[6], *dld[6], *dmw[6];
    for (int i = 0; i < 6; ++i) {
        if (i == 0) {  // keys: A,B,C,log_dt,mix_w
            dA[i] = F(ix++); dB[i] = F(ix++); dC[i] = F(ix++); dld[i] = F(ix++); dmw[i] = F(ix++);
            dcb[i] = dcw[i] = dlb[i] = dlg[i] = nullptr;
        } else {       // keys: A,B,C,conv_b,conv_w,ln_b,ln_g,log_dt,mix_w
            dA[i] = F(ix++); dB[i] = F(ix++); dC[i] = F(ix++);
            dcb[i] = F(ix++); dcw[i] = F(ix++);
            dlb[i] = F(ix++); dlg[i] = F(ix++);
            dld[i] = F(ix++); dmw[i] = F(ix++);
        }
    }
    const float *hA[2], *hB[2], *hC[2], *hcb[2], *hcw[2], *hlb[2], *hlg[2], *hld[2];
    for (int j = 0; j < 2; ++j) {  // keys: A,B,C,conv_b,conv_w,ln_b,ln_g,log_dt
        hA[j] = F(ix++); hB[j] = F(ix++); hC[j] = F(ix++);
        hcb[j] = F(ix++); hcw[j] = F(ix++);
        hlb[j] = F(ix++); hlg[j] = F(ix++); hld[j] = F(ix++);
    }
    const float *lA[2], *lB[2], *lC[2], *lld[2];
    for (int j = 0; j < 2; ++j) {  // keys: A,B,C,log_dt
        lA[j] = F(ix++); lB[j] = F(ix++); lC[j] = F(ix++); lld[j] = F(ix++);
    }
    // uda keys sorted: bk,bo,bq,bv,wk,wo,wq,wv
    const float* u_bk = F(ix++); const float* u_bo = F(ix++);
    const float* u_bq = F(ix++); const float* u_bv = F(ix++);
    const float* u_wk = F(ix++); const float* u_wo = F(ix++);
    const float* u_wq = F(ix++); const float* u_wv = F(ix++);
    const float *uA[6], *uB[6], *uC[6], *ucb[6], *ucw[6], *ulb[6], *ulg[6], *uld[6], *umw[6];
    for (int i = 0; i < 6; ++i) {
        if (i == 0) {
            uA[i] = F(ix++); uB[i] = F(ix++); uC[i] = F(ix++); uld[i] = F(ix++); umw[i] = F(ix++);
            ucb[i] = ucw[i] = ulb[i] = ulg[i] = nullptr;
        } else {
            uA[i] = F(ix++); uB[i] = F(ix++); uC[i] = F(ix++);
            ucb[i] = F(ix++); ucw[i] = F(ix++);
            ulb[i] = F(ix++); ulg[i] = F(ix++);
            uld[i] = F(ix++); umw[i] = F(ix++);
        }
    }

    // workspace carving
    char* wsb = (char*)d_ws;
    size_t off = 0;
    auto carve = [&](size_t nf) -> float* {
        float* p = (float*)(wsb + off);
        off += ((nf * sizeof(float) + 255) & ~(size_t)255);
        return p;
    };
    float* skipb[6];
    for (int i = 0; i < 6; ++i) skipb[i] = carve((size_t)Bb * CH[i] * Ls[i]);
    const size_t ACT = (size_t)1 << 21;
    float* A0 = carve(ACT); float* A1 = carve(ACT);
    float* A2 = carve(ACT); float* A3 = carve(ACT);
    float* CT = carve(ACT);
    float* seq   = carve((size_t)Bb * 65536);
    float* delta = carve((size_t)Bb * 65536);
    float* d100  = carve((size_t)Bb * 10000);
    float* qb = carve(40000); float* kb = carve(40000); float* vb = carve(40000);
    float* ab = carve(40000); float* ob = carve(40000);
    char* scr = wsb + off;

    // ---------------- down path ----------------
    const float* x = x_in;
    int c = 1, L = Ls[0];
    float* r0 = A0; float* r1 = A1; float* r2 = A2;
    for (int i = 0; i < 6; ++i) {
        hipMemcpyAsync(skipb[i], x, (size_t)Bb * c * L * sizeof(float),
                       hipMemcpyDeviceToDevice, stream);
        run_block(stream, scr, x, r0, CT, Bb, c, L, true,
                  dcw[i], dcb[i], dlg[i], dlb[i], dA[i], dB[i], dC[i], dld[i]);
        int cout = CH[i + 1], r = RS[i], Lo = L / r;
        mix_down<<<dim3((Lo + 255) / 256, cout, Bb), 256, 0, stream>>>(
            r0, dmw[i], r1, c, cout, Lo, r);
        x = r1;
        float* t = r0; r0 = r2; r2 = r1; r1 = t;
        c = cout; L = Lo;
    }

    // ---------------- bottleneck UDA ----------------
    int kdim = Ls[6];                   // 256
    long long tot = (long long)Bb * 256 * kdim;
    pack_seq<<<(int)((tot + 255) / 256), 256, 0, stream>>>(x, seq, Bb, kdim);
    interp1d_k<<<dim3((10000 + 255) / 256, Bb), 256, 0, stream>>>(seq, d100, 65536, 10000);
    {
        dim3 g(((400 + 15) / 16 + 7) / 8, (100 + 15) / 16);
        wmma_gemm_f16w<<<g, 256, 0, stream>>>(d100, u_wq, u_bq, qb, 400, 100, 100);
        wmma_gemm_f16w<<<g, 256, 0, stream>>>(vf,   u_wk, u_bk, kb, 400, 100, 100);
        wmma_gemm_f16w<<<g, 256, 0, stream>>>(vf,   u_wv, u_bv, vb, 400, 100, 100);
        mha_attn<<<dim3(100, 4, Bb), 32, 0, stream>>>(qb, kb, vb, ab, 100, 100, 4, 25);
        wmma_gemm_f16w<<<g, 256, 0, stream>>>(ab, u_wo, u_bo, ob, 400, 100, 100);
    }
    interp1d_k<<<dim3((65536 + 255) / 256, Bb), 256, 0, stream>>>(ob, delta, 10000, 65536);
    addv<<<(int)(((long long)Bb * 65536 + 255) / 256), 256, 0, stream>>>(
        seq, delta, seq, (long long)Bb * 65536);
    unpack_seq<<<(int)((tot + 255) / 256), 256, 0, stream>>>(seq, A3, Bb, kdim);

    // ---------------- hidden blocks ----------------
    run_block(stream, scr, A3, A0, CT, Bb, 256, kdim, true,
              hcw[0], hcb[0], hlg[0], hlb[0], hA[0], hB[0], hC[0], hld[0]);
    run_block(stream, scr, A0, A1, CT, Bb, 256, kdim, true,
              hcw[1], hcb[1], hlg[1], hlb[1], hA[1], hB[1], hC[1], hld[1]);
    x = A1; c = 256; L = kdim;

    // ---------------- up path ----------------
    r0 = A2; r1 = A3; r2 = A0;
    for (int j = 5; j >= 0; --j) {
        int cin = CH[j + 1], cout = CH[j], r = RS[j], Lo = L * r;
        mix_up_addskip<<<dim3((Lo + 255) / 256, cout, Bb), 256, 0, stream>>>(
            x, umw[j], skipb[j], r0, cin, cout, L, r);
        run_block(stream, scr, r0, r1, CT, Bb, cout, Lo, true,
                  ucw[j], ucb[j], ulg[j], ulb[j], uA[j], uB[j], uC[j], uld[j]);
        x = r1;
        float* t = r0; r0 = r2; r2 = r1; r1 = t;
        c = cout; L = Lo;
    }

    // ---------------- last blocks ----------------
    float* tlast = (x == A0) ? A1 : A0;
    run_block(stream, scr, x, tlast, CT, Bb, 1, 65536, true,
              nullptr, nullptr, nullptr, nullptr, lA[0], lB[0], lC[0], lld[0]);
    run_block(stream, scr, tlast, (float*)d_out, CT, Bb, 1, 65536, false,
              nullptr, nullptr, nullptr, nullptr, lA[1], lB[1], lC[1], lld[1]);
}
